// SingleStreamBlock_14139032338798
// MI455X (gfx1250) — matrix operand order repaired and checked by reading, not compiled, not run on hardware
//
#include <hip/hip_runtime.h>
#include <math.h>

// ---------------------------------------------------------------------------
// SingleStreamBlock (DiT) for MI455X / gfx1250, wave32 + WMMA bf16.
// B=1, L=2048, DM=3072, H=24, D=128, MLP=12288.
// GEMMs: double-buffered LDS pipeline fed by GLOBAL_LOAD_ASYNC_TO_LDS_B128.
// ---------------------------------------------------------------------------

typedef __attribute__((ext_vector_type(16))) __bf16 bf16x16;
typedef __attribute__((ext_vector_type(8)))  __bf16 bf16x8;
typedef __attribute__((ext_vector_type(8)))  float  f32x8;

#define DM    3072
#define LSEQ  2048
#define NH    24
#define HD    128
#define MLPD  12288
#define N1    (3*DM + MLPD)   /* 21504 */
#define N2CAT (DM + MLPD)     /* 15360 */

__device__ __forceinline__ f32x8 zero8() {
  f32x8 v;
#pragma unroll
  for (int i = 0; i < 8; i++) v[i] = 0.f;
  return v;
}

__device__ __forceinline__ bf16x16 join8(bf16x8 lo, bf16x8 hi) {
  bf16x16 r;
#pragma unroll
  for (int i = 0; i < 8; i++) { r[i] = lo[i]; r[8 + i] = hi[i]; }
  return r;
}

__device__ __forceinline__ f32x8 wmma_bf16(bf16x16 a, bf16x16 b, f32x8 c) {
  // D = A(16x32 bf16) * B(32x16 bf16) + C(16x16 f32)
  return __builtin_amdgcn_wmma_f32_16x16x32_bf16(false, a, false, b,
                                                 (short)0, c, false, false);
}

// Async DMA: 16B from global straight into LDS (no VGPR roundtrip, ASYNCcnt).
__device__ __forceinline__ void async_ld_b128(unsigned lds_off, const void* g) {
  asm volatile("global_load_async_to_lds_b128 %0, %1, off"
               :: "v"(lds_off), "v"((unsigned long long)(uintptr_t)g)
               : "memory");
}
__device__ __forceinline__ void wait_async0() {
  asm volatile("s_wait_asynccnt 0" ::: "memory");
}

__device__ __forceinline__ float xor_max16(float v) {
#pragma unroll
  for (int m = 1; m < 16; m <<= 1) v = fmaxf(v, __shfl_xor(v, m, 32));
  return v;
}
__device__ __forceinline__ float xor_sum16(float v) {
#pragma unroll
  for (int m = 1; m < 16; m <<= 1) v += __shfl_xor(v, m, 32);
  return v;
}

// ---------------------------------------------------------------------------
// 1) silu(vec)
// ---------------------------------------------------------------------------
__global__ void silu_vec_kernel(const float* __restrict__ vec, float* __restrict__ sv) {
  int i = blockIdx.x * blockDim.x + threadIdx.x;
  if (i < DM) { float v = vec[i]; sv[i] = v / (1.f + __expf(-v)); }
}

// ---------------------------------------------------------------------------
// 2) mod = silu(vec) @ mod_w + mod_b   (1x3072 @ 3072x9216 — tiny GEMV)
// ---------------------------------------------------------------------------
__global__ void mod_gemv_kernel(const float* __restrict__ sv,
                                const float* __restrict__ mw,
                                const float* __restrict__ mb,
                                float* __restrict__ mod) {
  int o = blockIdx.x * blockDim.x + threadIdx.x;
  if (o >= 3 * DM) return;
  float acc = 0.f;
  for (int i = 0; i < DM; i++) acc = fmaf(sv[i], mw[(size_t)i * (3 * DM) + o], acc);
  mod[o] = acc + mb[o];
}

// ---------------------------------------------------------------------------
// 3) Fused transpose + fp32->bf16:  src[R][C] f32  ->  dst[C][R] bf16
// ---------------------------------------------------------------------------
__global__ __launch_bounds__(256)
void transpose_f32_to_bf16_kernel(const float* __restrict__ src,
                                  __bf16* __restrict__ dst, int R, int C) {
  __shared__ float t[32][33];
  int tc = blockIdx.x * 32;
  int tr = blockIdx.y * 32;
  int lx = threadIdx.x & 31;
  int ly = threadIdx.x >> 5;
#pragma unroll
  for (int i = 0; i < 32; i += 8)
    t[ly + i][lx] = src[(size_t)(tr + ly + i) * C + tc + lx];
  __syncthreads();
#pragma unroll
  for (int i = 0; i < 32; i += 8)
    dst[(size_t)(tc + ly + i) * R + tr + lx] = (__bf16)t[lx][ly + i];
}

// ---------------------------------------------------------------------------
// 4) LayerNorm + modulation: x_mod = (1+scale)*xn + shift  -> bf16
// ---------------------------------------------------------------------------
__global__ __launch_bounds__(256)
void ln_mod_kernel(const float* __restrict__ x, const float* __restrict__ mod,
                   __bf16* __restrict__ xmod) {
  int row = blockIdx.x;
  const float* xr = x + (size_t)row * DM;
  __shared__ float r1[256], r2[256];
  float s = 0.f, s2 = 0.f;
  for (int c = threadIdx.x; c < DM; c += 256) { float v = xr[c]; s += v; s2 += v * v; }
  r1[threadIdx.x] = s; r2[threadIdx.x] = s2;
  __syncthreads();
  for (int off = 128; off > 0; off >>= 1) {
    if ((int)threadIdx.x < off) { r1[threadIdx.x] += r1[threadIdx.x + off];
                                  r2[threadIdx.x] += r2[threadIdx.x + off]; }
    __syncthreads();
  }
  float mu  = r1[0] * (1.f / DM);
  float var = r2[0] * (1.f / DM) - mu * mu;
  float rr  = rsqrtf(var + 1e-6f);
  for (int c = threadIdx.x; c < DM; c += 256) {
    float xn = (xr[c] - mu) * rr;
    float xm = (1.f + mod[DM + c]) * xn + mod[c];
    xmod[(size_t)row * DM + c] = (__bf16)xm;
  }
}

// ---------------------------------------------------------------------------
// 5,9) Tiled WMMA GEMM:  out[M,N] = A[M,K](bf16) @ W[K,N] + bias,
//      W supplied TRANSPOSED: Bt[N][K] bf16.
//      Double-buffered LDS, async global->LDS DMA, 8 WMMAs/wave/k-step.
// ---------------------------------------------------------------------------
__global__ __launch_bounds__(256)
void gemm_wmma_kernel(const __bf16* __restrict__ A, const __bf16* __restrict__ Bt,
                      const float* __restrict__ bias, float* __restrict__ out,
                      const float* __restrict__ resid, const float* __restrict__ gate,
                      int M, int N, int K, int fuse_residual) {
  __shared__ __bf16 As[2][128][48];   // ping-pong: 128 rows x 32 k, stride 48
  __shared__ __bf16 Bs[2][128][48];   // ping-pong: 128 cols x 32 k

  const int tid = threadIdx.x;
  const int nb = blockIdx.x, mb = blockIdx.y;
  const int wid = tid >> 5, lane = tid & 31;
  const int wm = wid >> 2, wn = wid & 3;
  const int lrow = lane & 15, kh = lane >> 4;

  // Staging: each thread owns 2 rows of A-tile and 2 rows of Bt-tile (16B each).
  const int sr = tid >> 2;            // 0..63
  const int sp = (tid & 3) * 8;       // element offset within 32-k window
  const __bf16* Ag = A  + (size_t)(mb * 128 + sr) * K + sp;
  const __bf16* Bg = Bt + (size_t)(nb * 128 + sr) * K + sp;
  const size_t rowskip = (size_t)64 * K;

  const unsigned aOff0[2] = { (unsigned)(uintptr_t)&As[0][sr][sp],
                              (unsigned)(uintptr_t)&As[1][sr][sp] };
  const unsigned aOff1[2] = { (unsigned)(uintptr_t)&As[0][sr + 64][sp],
                              (unsigned)(uintptr_t)&As[1][sr + 64][sp] };
  const unsigned bOff0[2] = { (unsigned)(uintptr_t)&Bs[0][sr][sp],
                              (unsigned)(uintptr_t)&Bs[1][sr][sp] };
  const unsigned bOff1[2] = { (unsigned)(uintptr_t)&Bs[0][sr + 64][sp],
                              (unsigned)(uintptr_t)&Bs[1][sr + 64][sp] };

  auto issue = [&](int buf, int kt) {
    const size_t ko = (size_t)kt * 32;
    async_ld_b128(aOff0[buf], Ag + ko);
    async_ld_b128(aOff1[buf], Ag + rowskip + ko);
    async_ld_b128(bOff0[buf], Bg + ko);
    async_ld_b128(bOff1[buf], Bg + rowskip + ko);
  };

  f32x8 acc[4][2];
#pragma unroll
  for (int tm = 0; tm < 4; tm++)
#pragma unroll
    for (int tn = 0; tn < 2; tn++) acc[tm][tn] = zero8();

  const int ksteps = K >> 5;
  issue(0, 0);
  int cur = 0;
  for (int i = 0; i < ksteps; ++i) {
    wait_async0();        // this wave's DMAs into LDS have landed
    __syncthreads();      // everyone's have
    if (i + 1 < ksteps) issue(cur ^ 1, i + 1);   // prefetch next tile (other buffer)

    // A fragment: lane M=lrow; K = kh*8+0..7 and 16+kh*8+0..7
    bf16x16 af[4];
#pragma unroll
    for (int tm = 0; tm < 4; tm++) {
      const __bf16* pA = &As[cur][wm * 64 + tm * 16 + lrow][kh * 8];
      af[tm] = join8(*(const bf16x8*)pA, *(const bf16x8*)(pA + 16));
    }
    // B fragment: lane col=lrow; K = kh*8+0..7 and 16+kh*8+0..7
    bf16x16 bfg[2];
#pragma unroll
    for (int tn = 0; tn < 2; tn++) {
      const __bf16* pB = &Bs[cur][wn * 32 + tn * 16 + lrow][kh * 8];
      bfg[tn] = join8(*(const bf16x8*)pB, *(const bf16x8*)(pB + 16));
    }
#pragma unroll
    for (int tm = 0; tm < 4; tm++)
#pragma unroll
      for (int tn = 0; tn < 2; tn++)
        acc[tm][tn] = wmma_bf16(af[tm], bfg[tn], acc[tm][tn]);
    cur ^= 1;
  }

  // Epilogue. C layout: VGPR r, lane: M = r + 8*kh, N = lrow
#pragma unroll
  for (int tm = 0; tm < 4; tm++) {
#pragma unroll
    for (int tn = 0; tn < 2; tn++) {
      int col = nb * 128 + wn * 32 + tn * 16 + lrow;
      float bb = bias[col];
      float gg = fuse_residual ? gate[col] : 0.f;
#pragma unroll
      for (int r = 0; r < 8; r++) {
        int row = mb * 128 + wm * 64 + tm * 16 + r + 8 * kh;
        float v = acc[tm][tn][r] + bb;
        if (fuse_residual) v = resid[(size_t)row * N + col] + gg * v;
        out[(size_t)row * N + col] = v;
      }
    }
  }
}

// ---------------------------------------------------------------------------
// 6) qkv split + RMS-norm(q,k) + rope(q,k).
//    Q, K head-major [h][l][d]; V head-major d-major [h][d][l].
// ---------------------------------------------------------------------------
__global__ __launch_bounds__(128)
void qkv_prep_kernel(const float* __restrict__ hbuf, const float* __restrict__ pe,
                     const float* __restrict__ qscale, const float* __restrict__ kscale,
                     __bf16* __restrict__ qb, __bf16* __restrict__ kb,
                     __bf16* __restrict__ vt) {
  int l = blockIdx.x, head = blockIdx.y, t = blockIdx.z;
  int d = threadIdx.x;
  float v = hbuf[(size_t)l * N1 + t * DM + head * HD + d];
  if (t == 2) {
    vt[((size_t)head * HD + d) * LSEQ + l] = (__bf16)v;   // [h][d][l]
    return;
  }
  __shared__ float red[128];
  __shared__ float tn[128];
  red[d] = v * v;
  __syncthreads();
  for (int off = 64; off > 0; off >>= 1) {
    if (d < off) red[d] += red[d + off];
    __syncthreads();
  }
  float rr = rsqrtf(red[0] * (1.f / HD) + 1e-6f);
  float scv = (t == 0) ? qscale[d] : kscale[d];
  tn[d] = v * rr * scv;
  __syncthreads();
  int j = d >> 1, rbit = d & 1;
  const float* pp = pe + ((((size_t)l * 64 + j) * 2 + rbit) * 2);
  float o = pp[0] * tn[2 * j] + pp[1] * tn[2 * j + 1];
  __bf16* dst = (t == 0) ? qb : kb;
  dst[((size_t)head * LSEQ + l) * HD + d] = (__bf16)o;
}

// ---------------------------------------------------------------------------
// 7) Flash attention, one wave per (head, 16-query tile).
// ---------------------------------------------------------------------------
__global__ __launch_bounds__(32)
void attn_kernel(const __bf16* __restrict__ Qb, const __bf16* __restrict__ Kb,
                 const __bf16* __restrict__ Vtb, __bf16* __restrict__ a2) {
  const int qt = blockIdx.x;
  const int h  = blockIdx.y;
  const __bf16* Q  = Qb  + (size_t)h * LSEQ * HD;
  const __bf16* Km = Kb  + (size_t)h * LSEQ * HD;
  const __bf16* Vt = Vtb + (size_t)h * HD * LSEQ;   // [d][l]
  const int lane = threadIdx.x, lrow = lane & 15, kh = lane >> 4;

  __shared__ __bf16 Pl[16][32];

  bf16x16 qf[4];
  const int qrow = qt * 16 + lrow;
#pragma unroll
  for (int kc = 0; kc < 4; kc++) {
    const __bf16* p = Q + (size_t)qrow * HD + kc * 32 + kh * 8;
    qf[kc] = join8(*(const bf16x8*)p, *(const bf16x8*)(p + 16));
  }

  f32x8 o[8];
#pragma unroll
  for (int t = 0; t < 8; t++) o[t] = zero8();
  float m[8], lsum[8];
#pragma unroll
  for (int r = 0; r < 8; r++) { m[r] = -1e30f; lsum[r] = 0.f; }
  const float sc = 0.08838834764831845f;  // 1/sqrt(128)

  for (int kb = 0; kb < LSEQ; kb += 32) {
    f32x8 s0 = zero8(), s1 = zero8();
#pragma unroll
    for (int kc = 0; kc < 4; kc++) {
      const __bf16* p0 = Km + (size_t)(kb + lrow) * HD + kc * 32 + kh * 8;
      bf16x16 kf0 = join8(*(const bf16x8*)p0, *(const bf16x8*)(p0 + 16));
      s0 = wmma_bf16(qf[kc], kf0, s0);
      const __bf16* p1 = Km + (size_t)(kb + 16 + lrow) * HD + kc * 32 + kh * 8;
      bf16x16 kf1 = join8(*(const bf16x8*)p1, *(const bf16x8*)(p1 + 16));
      s1 = wmma_bf16(qf[kc], kf1, s1);
    }

    float p0v[8], p1v[8];
#pragma unroll
    for (int r = 0; r < 8; r++) {
      float a = fmaxf(s0[r], s1[r]) * sc;
      a = xor_max16(a);
      float mn = fmaxf(m[r], a);
      float corr = __expf(m[r] - mn);
      m[r] = mn;
      p0v[r] = __expf(s0[r] * sc - mn);
      p1v[r] = __expf(s1[r] * sc - mn);
      float rs = xor_sum16(p0v[r] + p1v[r]);
      lsum[r] = lsum[r] * corr + rs;
#pragma unroll
      for (int t = 0; t < 8; t++) o[t][r] *= corr;
    }

    __syncthreads();
#pragma unroll
    for (int r = 0; r < 8; r++) {
      Pl[r + 8 * kh][lrow]      = (__bf16)p0v[r];
      Pl[r + 8 * kh][lrow + 16] = (__bf16)p1v[r];
    }
    __syncthreads();
    const __bf16* pp = &Pl[lrow][kh * 8];
    bf16x16 pf = join8(*(const bf16x8*)pp, *(const bf16x8*)(pp + 16));

#pragma unroll
    for (int dt = 0; dt < 8; dt++) {
      const __bf16* pv = Vt + (size_t)(dt * 16 + lrow) * LSEQ + kb + kh * 8;
      bf16x16 vf = join8(*(const bf16x8*)pv, *(const bf16x8*)(pv + 16));
      o[dt] = wmma_bf16(pf, vf, o[dt]);
    }
  }

#pragma unroll
  for (int r = 0; r < 8; r++) lsum[r] = 1.f / lsum[r];
#pragma unroll
  for (int dt = 0; dt < 8; dt++) {
    int col = h * HD + dt * 16 + lrow;
#pragma unroll
    for (int r = 0; r < 8; r++) {
      int row = qt * 16 + r + 8 * kh;
      a2[(size_t)row * N2CAT + col] = (__bf16)(o[dt][r] * lsum[r]);
    }
  }
}

// ---------------------------------------------------------------------------
// 8) gelu(tanh) of mlp half -> bf16 into concat buffer
// ---------------------------------------------------------------------------
__global__ void gelu_kernel(const float* __restrict__ hbuf, __bf16* __restrict__ a2) {
  const long n = (long)LSEQ * MLPD;
  long stride = (long)gridDim.x * blockDim.x;
  for (long i = (long)blockIdx.x * blockDim.x + threadIdx.x; i < n; i += stride) {
    long row = i / MLPD, j = i % MLPD;
    float v = hbuf[row * (long)N1 + 3 * DM + j];
    float g = 0.5f * v * (1.f + tanhf(0.7978845608028654f * (v + 0.044715f * v * v * v)));
    a2[row * (long)N2CAT + DM + j] = (__bf16)g;
  }
}

// ---------------------------------------------------------------------------
// Host launcher
// ---------------------------------------------------------------------------
extern "C" void kernel_launch(void* const* d_in, const int* in_sizes, int n_in,
                              void* d_out, int out_size, void* d_ws, size_t ws_size,
                              hipStream_t stream) {
  (void)in_sizes; (void)n_in; (void)out_size; (void)ws_size;
  const float* x    = (const float*)d_in[0];
  const float* vec  = (const float*)d_in[1];
  const float* pe   = (const float*)d_in[2];
  const float* modw = (const float*)d_in[3];
  const float* modb = (const float*)d_in[4];
  const float* w1   = (const float*)d_in[5];
  const float* b1   = (const float*)d_in[6];
  const float* w2   = (const float*)d_in[7];
  const float* b2   = (const float*)d_in[8];
  const float* qs   = (const float*)d_in[9];
  const float* ks   = (const float*)d_in[10];
  float* out = (float*)d_out;

  char* wsp = (char*)d_ws;
  size_t off = 0;
  auto carve = [&](size_t bytes) -> char* {
    char* p = wsp + off;
    off += (bytes + 255) & ~(size_t)255;
    return p;
  };
  float*  sv   = (float*)carve((size_t)DM * 4);
  float*  modv = (float*)carve((size_t)3 * DM * 4);
  __bf16* xmod = (__bf16*)carve((size_t)LSEQ * DM * 2);
  __bf16* w1t  = (__bf16*)carve((size_t)DM * N1 * 2);      // [N1][DM]
  __bf16* w2t  = (__bf16*)carve((size_t)N2CAT * DM * 2);   // [DM][N2CAT]
  float*  hbuf = (float*)carve((size_t)LSEQ * N1 * 4);
  __bf16* qb   = (__bf16*)carve((size_t)NH * LSEQ * HD * 2);
  __bf16* kb   = (__bf16*)carve((size_t)NH * LSEQ * HD * 2);
  __bf16* vt   = (__bf16*)carve((size_t)NH * HD * LSEQ * 2);
  __bf16* a2   = (__bf16*)carve((size_t)LSEQ * N2CAT * 2);

  silu_vec_kernel<<<(DM + 255) / 256, 256, 0, stream>>>(vec, sv);
  mod_gemv_kernel<<<(3 * DM + 255) / 256, 256, 0, stream>>>(sv, modw, modb, modv);

  // W1[3072][21504] -> W1t[21504][3072] bf16 ; W2[15360][3072] -> W2t[3072][15360]
  transpose_f32_to_bf16_kernel<<<dim3(N1 / 32, DM / 32), 256, 0, stream>>>(w1, w1t, DM, N1);
  transpose_f32_to_bf16_kernel<<<dim3(DM / 32, N2CAT / 32), 256, 0, stream>>>(w2, w2t, N2CAT, DM);

  ln_mod_kernel<<<LSEQ, 256, 0, stream>>>(x, modv, xmod);

  // h = x_mod @ w1 + b1   (2048 x 21504, K=3072)
  gemm_wmma_kernel<<<dim3(N1 / 128, LSEQ / 128), 256, 0, stream>>>(
      xmod, w1t, b1, hbuf, nullptr, nullptr, LSEQ, N1, DM, 0);

  qkv_prep_kernel<<<dim3(LSEQ, NH, 3), 128, 0, stream>>>(hbuf, pe, qs, ks, qb, kb, vt);
  attn_kernel<<<dim3(LSEQ / 16, NH), 32, 0, stream>>>(qb, kb, vt, a2);
  gelu_kernel<<<4096, 256, 0, stream>>>(hbuf, a2);

  // out = x + gate * (a2 @ w2 + b2)   (2048 x 3072, K=15360)
  gemm_wmma_kernel<<<dim3(DM / 128, LSEQ / 128), 256, 0, stream>>>(
      a2, w2t, b2, out, x, modv + 2 * DM, LSEQ, DM, N2CAT, 1);
}
